// GCNSubModule_89876485636648
// MI455X (gfx1250) — matrix operand order repaired and checked by reading, not compiled, not run on hardware
//
#include <hip/hip_runtime.h>

typedef __attribute__((ext_vector_type(16))) _Float16 v16h;
typedef __attribute__((ext_vector_type(8)))  float    v8f;

#define D 64
#define EPSV 1e-5f

// ---------------- init: zero out, deg=1 (self loop), zero stats ----------------
__global__ void gcn_init_kernel(float* __restrict__ out, float* __restrict__ deg,
                                float* __restrict__ stats, int n) {
  int idx = blockIdx.x * blockDim.x + threadIdx.x;
  if (idx < n * D) out[idx] = 0.0f;
  if (idx < n) deg[idx] = 1.0f;          // self-loop contribution
  if (idx < 2 * D) stats[idx] = 0.0f;
}

// ---------------- degree accumulation over edge targets ----------------
__global__ void gcn_degree_kernel(const int* __restrict__ dst, float* __restrict__ deg, int E) {
  int e = blockIdx.x * blockDim.x + threadIdx.x;
  if (e < E) unsafeAtomicAdd(&deg[dst[e]], 1.0f);
}

// ---------------- deg -> dinv in place ----------------
__global__ void gcn_dinv_kernel(float* __restrict__ deg, int n) {
  int i = blockIdx.x * blockDim.x + threadIdx.x;
  if (i < n) {
    float d = deg[i];
    deg[i] = d > 0.0f ? rsqrtf(d) : 0.0f;
  }
}

// ---------------- pre-pack W (f32 row-major 64x64) into f16 WMMA B fragments --------
// Fragment layout: wpack[((colTile*2 + kblock)*32 + lane)*16 + idx]
//   B frag elem idx -> K = kblock*32 + (idx>>3)*16 + (lane>>4)*8 + (idx&7), col = colTile*16 + (lane&15)
// Total: 4 colTiles x 2 kblocks x 32 lanes x 16 f16 = 4096 elems = 8 KB (stays hot in L1/L2).
__global__ void gcn_packW_kernel(const float* __restrict__ W, _Float16* __restrict__ wpack) {
  int tid = blockIdx.x * blockDim.x + threadIdx.x;
  if (tid >= 4096) return;
  int idx     = tid & 15;
  int lane    = (tid >> 4) & 31;
  int kblock  = (tid >> 9) & 1;
  int colTile = tid >> 10;
  int k   = kblock * 32 + (idx >> 3) * 16 + (lane >> 4) * 8 + (idx & 7);
  int col = colTile * 16 + (lane & 15);
  wpack[tid] = (_Float16)W[k * D + col];
}

// ---------------- h = x @ W via v_wmma_f32_16x16x32_f16 ----------------
// One wave32 computes a 16x16 output tile; K=64 split into two K=32 WMMA ops.
// Block = 256 threads = 8 waves = 2 row-tiles x 4 col-tiles.
__global__ void __launch_bounds__(256) gcn_gemm_wmma_kernel(
    const float* __restrict__ x, const _Float16* __restrict__ wpack,
    float* __restrict__ h, int n) {
  const int lane = threadIdx.x & 31;
  const int wave = threadIdx.x >> 5;                 // 0..7
  const int rowBase = (blockIdx.x * 2 + (wave >> 2)) * 16;
  const int colTile = wave & 3;                      // D=64 -> 4 col tiles
  const int colBase = colTile * 16;
  if (rowBase >= n) return;                          // wave-uniform guard (EXEC stays full)
  const int half = lane >> 4;                        // 0 or 1
  const int m = lane & 15;

  int arow = rowBase + m;                            // clamp only matters for tail tiles
  if (arow >= n) arow = n - 1;
  const float4* xr4 = (const float4*)(x + (size_t)arow * D);

  // A fragment (16-bit A 16x32 layout): elems 0..7 -> K = K0 + half*8 + t,
  //                                     elems 8..15 -> K = K0 + 16 + half*8 + t
  // Contiguous 8-float runs -> float4 vector loads (global_load_b128).
  float4 q0 = xr4[half * 2 + 0],  q1 = xr4[half * 2 + 1];       // K =      half*8 + 0..7
  float4 q2 = xr4[4 + half * 2],  q3 = xr4[5 + half * 2];       // K = 16 + half*8 + 0..7
  float4 q4 = xr4[8 + half * 2],  q5 = xr4[9 + half * 2];       // K = 32 + half*8 + 0..7
  float4 q6 = xr4[12 + half * 2], q7 = xr4[13 + half * 2];      // K = 48 + half*8 + 0..7

  v16h a0, a1;
  a0[0]  = (_Float16)q0.x; a0[1]  = (_Float16)q0.y; a0[2]  = (_Float16)q0.z; a0[3]  = (_Float16)q0.w;
  a0[4]  = (_Float16)q1.x; a0[5]  = (_Float16)q1.y; a0[6]  = (_Float16)q1.z; a0[7]  = (_Float16)q1.w;
  a0[8]  = (_Float16)q2.x; a0[9]  = (_Float16)q2.y; a0[10] = (_Float16)q2.z; a0[11] = (_Float16)q2.w;
  a0[12] = (_Float16)q3.x; a0[13] = (_Float16)q3.y; a0[14] = (_Float16)q3.z; a0[15] = (_Float16)q3.w;
  a1[0]  = (_Float16)q4.x; a1[1]  = (_Float16)q4.y; a1[2]  = (_Float16)q4.z; a1[3]  = (_Float16)q4.w;
  a1[4]  = (_Float16)q5.x; a1[5]  = (_Float16)q5.y; a1[6]  = (_Float16)q5.z; a1[7]  = (_Float16)q5.w;
  a1[8]  = (_Float16)q6.x; a1[9]  = (_Float16)q6.y; a1[10] = (_Float16)q6.z; a1[11] = (_Float16)q6.w;
  a1[12] = (_Float16)q7.x; a1[13] = (_Float16)q7.y; a1[14] = (_Float16)q7.z; a1[15] = (_Float16)q7.w;

  // B fragments: pre-packed, contiguous 32B per lane -> two b128 loads each.
  const v16h* wp = (const v16h*)wpack;
  v16h b0 = wp[(colTile * 2 + 0) * 32 + lane];
  v16h b1 = wp[(colTile * 2 + 1) * 32 + lane];

  v8f acc = {};
  acc = __builtin_amdgcn_wmma_f32_16x16x32_f16(false, a0, false, b0, (short)0, acc, false, false);
  acc = __builtin_amdgcn_wmma_f32_16x16x32_f16(false, a1, false, b1, (short)0, acc, false, false);

  // C/D layout: VGPR r -> row rowBase + r + 8*half, col = colBase + (lane&15)
  const int outRow = rowBase + half * 8;
  float* hp = h + (size_t)outRow * D + colBase + m;
  if (rowBase + 16 <= n) {                           // wave-uniform fast path: no exec dance
#pragma unroll
    for (int r = 0; r < 8; ++r) hp[(size_t)r * D] = acc[r];
  } else {                                           // cold tail path (unused for N%16==0)
#pragma unroll
    for (int r = 0; r < 8; ++r)
      if (outRow + r < n) hp[(size_t)r * D] = acc[r];
  }
}

// ---------------- normalized message scatter-add (edges + self loops) ----------------
// One wave per message; each lane adds 2 of the 64 features with native f32 atomics.
__global__ void gcn_scatter_kernel(const int* __restrict__ src, const int* __restrict__ dst,
                                   const float* __restrict__ h, const float* __restrict__ dinv,
                                   float* __restrict__ out, int E, int n) {
  int e = blockIdx.x * (blockDim.x >> 5) + (threadIdx.x >> 5);
  int lane = threadIdx.x & 31;
  if (e >= E + n) return;
  int s, d;
  if (e < E) { s = src[e]; d = dst[e]; }
  else       { s = e - E;  d = s; }                  // self loop
  float nrm = dinv[s] * dinv[d];
  const float* hs = h + (size_t)s * D;
  float* od = out + (size_t)d * D;
  unsafeAtomicAdd(&od[lane],      hs[lane]      * nrm);
  unsafeAtomicAdd(&od[lane + 32], hs[lane + 32] * nrm);
}

// ---------------- per-feature sum and sum-of-squares ----------------
__global__ void __launch_bounds__(256) gcn_stats_kernel(const float* __restrict__ out,
                                                        float* __restrict__ stats, int n) {
  __shared__ float red[2][256];
  int col = threadIdx.x & (D - 1);
  int sub = threadIdx.x >> 6;                         // 0..3 row sub-lanes per column
  float s = 0.f, s2 = 0.f;
  for (int r = blockIdx.x * 4 + sub; r < n; r += gridDim.x * 4) {
    float v = out[(size_t)r * D + col];
    s += v; s2 += v * v;
  }
  red[0][threadIdx.x] = s;
  red[1][threadIdx.x] = s2;
  __syncthreads();
  if (sub == 0) {
    s  = red[0][col] + red[0][col + 64] + red[0][col + 128] + red[0][col + 192];
    s2 = red[1][col] + red[1][col + 64] + red[1][col + 128] + red[1][col + 192];
    unsafeAtomicAdd(&stats[col], s);
    unsafeAtomicAdd(&stats[D + col], s2);
  }
}

// ---------------- batch-norm (batch stats) + affine + ReLU, in place ----------------
// Note: the conv bias b cancels exactly in (out - mean), so it is never applied.
__global__ void gcn_finalize_kernel(float* __restrict__ out, const float* __restrict__ stats,
                                    const float* __restrict__ gamma, const float* __restrict__ beta,
                                    int n) {
  int idx = blockIdx.x * blockDim.x + threadIdx.x;
  if (idx >= n * D) return;
  int col = idx & (D - 1);
  float invn = 1.0f / (float)n;
  float mean = stats[col] * invn;
  float var  = stats[D + col] * invn - mean * mean;
  float scale = gamma[col] * rsqrtf(var + EPSV);
  float v = (out[idx] - mean) * scale + beta[col];
  out[idx] = v > 0.0f ? v : 0.0f;
}

extern "C" void kernel_launch(void* const* d_in, const int* in_sizes, int n_in,
                              void* d_out, int out_size, void* d_ws, size_t ws_size,
                              hipStream_t stream) {
  const float* x     = (const float*)d_in[0];
  const int*   ei    = (const int*)d_in[1];
  const float* W     = (const float*)d_in[2];
  // d_in[3] = b: mathematically cancels under batch-norm with batch statistics.
  const float* gamma = (const float*)d_in[4];
  const float* beta  = (const float*)d_in[5];

  const int n = in_sizes[0] / D;     // 100000
  const int E = in_sizes[1] / 2;     // 1600000
  const int* src = ei;
  const int* dst = ei + E;

  float* ws       = (float*)d_ws;
  float* deg      = ws;                      // n floats; becomes dinv in place
  float* stats    = ws + n;                  // 2*D floats (sum, sumsq)
  float* h        = stats + 2 * D;           // n*D floats
  _Float16* wpack = (_Float16*)(h + (size_t)n * D);  // 4096 f16 (8 KB), 32B-aligned
  float* out      = (float*)d_out;

  const int totalND = n * D;
  gcn_init_kernel<<<(totalND + 255) / 256, 256, 0, stream>>>(out, deg, stats, n);
  gcn_packW_kernel<<<16, 256, 0, stream>>>(W, wpack);
  gcn_degree_kernel<<<(E + 255) / 256, 256, 0, stream>>>(dst, deg, E);
  gcn_dinv_kernel<<<(n + 255) / 256, 256, 0, stream>>>(deg, n);

  const int rowTiles = (n + 15) / 16;
  gcn_gemm_wmma_kernel<<<(rowTiles + 1) / 2, 256, 0, stream>>>(x, wpack, h, n);

  const int totalMsgs = E + n;
  gcn_scatter_kernel<<<(totalMsgs + 7) / 8, 256, 0, stream>>>(src, dst, h, deg, out, E, n);

  gcn_stats_kernel<<<256, 256, 0, stream>>>(out, stats, n);
  gcn_finalize_kernel<<<(totalND + 255) / 256, 256, 0, stream>>>(out, stats, gamma, beta, n);
}
